// GATConv_88244398064423
// MI455X (gfx1250) — matrix operand order repaired and checked by reading, not compiled, not run on hardware
//
#include <hip/hip_runtime.h>
#include <math.h>

typedef __attribute__((ext_vector_type(16))) __bf16 v16bf;
typedef __attribute__((ext_vector_type(8)))  __bf16 v8bf;
typedef __attribute__((ext_vector_type(8)))  float  v8f;

#define H4 4
#define INDIM 128
#define EDGEDIM 16
#define OUTH 64
#define FMLP 272
#define FPAD 288        // 272 padded to 9*32 for bf16 WMMA K-steps
#define LDS1_STRIDE 296 // padded LDS row stride (bank-conflict-free column reads)
#define LDS2_STRIDE 72

// ---------- helpers ----------
__device__ __forceinline__ unsigned fkey(float f) {
    unsigned u = __float_as_uint(f);
    return (u & 0x80000000u) ? ~u : (u | 0x80000000u);
}
__device__ __forceinline__ float funkey(unsigned k) {
    return (k & 0x80000000u) ? __uint_as_float(k ^ 0x80000000u) : __uint_as_float(~k);
}

// ---------- 0: init m-keys, softmax sums, BN accumulators ----------
__global__ void init_kernel(unsigned* __restrict__ mkey, float* __restrict__ s,
                            float* __restrict__ bnbuf, int nh) {
    int i = blockIdx.x * 256 + threadIdx.x;
    if (i < nh) { mkey[i] = 0u; s[i] = 0.f; }
    if (i < 512) bnbuf[i] = 0.f;   // bn_sum[256] + bn_sq[256]
}

// ---------- 1: per-node attention logits (x @ a1^T, x @ a2^T), wave per node ----------
__global__ __launch_bounds__(256) void node_logits_kernel(
    const float* __restrict__ x, const float* __restrict__ a1w,
    const float* __restrict__ a2w, float* __restrict__ attn1,
    float* __restrict__ attn2, int N) {
    int lane = threadIdx.x & 31;
    int n = blockIdx.x * 8 + (threadIdx.x >> 5);
    if (n >= N) return;
    float4 xv = ((const float4*)(x + (size_t)n * INDIM))[lane];
#pragma unroll
    for (int h = 0; h < H4; ++h) {
        float4 w1v = ((const float4*)(a1w + h * INDIM))[lane];
        float4 w2v = ((const float4*)(a2w + h * INDIM))[lane];
        float d1 = xv.x * w1v.x + xv.y * w1v.y + xv.z * w1v.z + xv.w * w1v.w;
        float d2 = xv.x * w2v.x + xv.y * w2v.y + xv.z * w2v.z + xv.w * w2v.w;
#pragma unroll
        for (int off = 16; off; off >>= 1) {
            d1 += __shfl_xor(d1, off, 32);
            d2 += __shfl_xor(d2, off, 32);
        }
        if (lane == 0) { attn1[n * H4 + h] = d1; attn2[n * H4 + h] = d2; }
    }
}

// ---------- 2: pack w1 (pad 272->288) and w2 to bf16 ----------
__global__ void pack_weights_kernel(const float* __restrict__ w1,
                                    const float* __restrict__ w2,
                                    __bf16* __restrict__ w1bf,
                                    __bf16* __restrict__ w2bf) {
    int i = blockIdx.x * 256 + threadIdx.x;
    const int W1T = H4 * OUTH * FPAD;
    if (i < W1T) {
        int f = i % FPAD, ho = i / FPAD;
        w1bf[i] = (f < FMLP) ? (__bf16)w1[(size_t)ho * FMLP + f] : (__bf16)0.f;
    } else if (i - W1T < H4 * OUTH * OUTH) {
        int k = i - W1T;
        w2bf[k] = (__bf16)w2[k];
    }
}

// ---------- 3: edge logits + segment max (ordered-uint atomicMax) ----------
__global__ void edge_logits_kernel(const int* __restrict__ row, const int* __restrict__ col,
                                   const float* __restrict__ edge_attr,
                                   const float* __restrict__ aew,
                                   const float* __restrict__ attn1,
                                   const float* __restrict__ attn2,
                                   float* __restrict__ exbuf, unsigned* __restrict__ mkey,
                                   int E) {
    int e = blockIdx.x * 256 + threadIdx.x;
    if (e >= E) return;
    int r = row[e], c = col[e];
    float4 A1 = ((const float4*)attn1)[r];
    float4 A2 = ((const float4*)attn2)[c];
    float a1a[4] = {A1.x, A1.y, A1.z, A1.w};
    float a2a[4] = {A2.x, A2.y, A2.z, A2.w};
    float ea[16];
    const float4* ep = (const float4*)(edge_attr + (size_t)e * EDGEDIM);
#pragma unroll
    for (int q = 0; q < 4; ++q) {
        float4 v = ep[q];
        ea[4 * q] = v.x; ea[4 * q + 1] = v.y; ea[4 * q + 2] = v.z; ea[4 * q + 3] = v.w;
    }
    float lg[4];
#pragma unroll
    for (int h = 0; h < H4; ++h) {
        const float* w = aew + h * EDGEDIM;
        float d = 0.f;
#pragma unroll
        for (int k = 0; k < EDGEDIM; ++k) d += ea[k] * w[k];
        lg[h] = (a1a[h] + a2a[h] + d) * 0.125f;   // 1/sqrt(64)
        atomicMax(&mkey[r * H4 + h], fkey(lg[h]));
    }
    ((float4*)exbuf)[e] = make_float4(lg[0], lg[1], lg[2], lg[3]);
}

// ---------- 4: exp(logit - max) + segment sum ----------
__global__ void edge_exp_kernel(const int* __restrict__ row, float* __restrict__ exbuf,
                                const unsigned* __restrict__ mkey, float* __restrict__ s,
                                int E) {
    int e = blockIdx.x * 256 + threadIdx.x;
    if (e >= E) return;
    int r = row[e];
    float4 lg = ((const float4*)exbuf)[e];
    uint4 mk = ((const uint4*)mkey)[r];
    float4 ev;
    ev.x = __expf(lg.x - funkey(mk.x));
    ev.y = __expf(lg.y - funkey(mk.y));
    ev.z = __expf(lg.z - funkey(mk.z));
    ev.w = __expf(lg.w - funkey(mk.w));
    ((float4*)exbuf)[e] = ev;
    atomicAdd(&s[r * H4 + 0], ev.x);
    atomicAdd(&s[r * H4 + 1], ev.y);
    atomicAdd(&s[r * H4 + 2], ev.z);
    atomicAdd(&s[r * H4 + 3], ev.w);
}

// ---------- 5: CSR row_ptr via binary search over sorted row ----------
__global__ void row_ptr_kernel(const int* __restrict__ row, int* __restrict__ rp,
                               int N, int E) {
    int n = blockIdx.x * 256 + threadIdx.x;
    if (n > N) return;
    if (n == N) { rp[N] = E; return; }
    int lo = 0, hi = E;
    while (lo < hi) {
        int mid = (lo + hi) >> 1;
        if (row[mid] < n) lo = mid + 1; else hi = mid;
    }
    rp[n] = lo;
}

// ---------- 6: per-node aggregation + bf16 feature assembly [N][H][288] ----------
__global__ __launch_bounds__(256) void aggregate_kernel(
    const float* __restrict__ x, const int* __restrict__ col,
    const float* __restrict__ edge_attr, const float* __restrict__ exbuf,
    const float* __restrict__ s, const int* __restrict__ rp,
    __bf16* __restrict__ feats, int N) {
    int n = blockIdx.x;
    if (n >= N) return;
    int tid = threadIdx.x;
    int h = tid >> 6;     // head
    int j = tid & 63;     // dim-pair index
    float inv_s = 1.f / (s[n * H4 + h] + 1e-16f);
    int e0 = rp[n], e1 = rp[n + 1];
    float acc0 = 0.f, acc1 = 0.f, accE = 0.f;
    for (int e = e0; e < e1; ++e) {
        float a = exbuf[(size_t)e * H4 + h] * inv_s;
        int c = col[e];
        if (e + 1 < e1) __builtin_prefetch(x + (size_t)col[e + 1] * INDIM, 0, 3);
        float2 xv = ((const float2*)x)[(size_t)c * 64 + j];
        acc0 += a * xv.x;
        acc1 += a * xv.y;
        if (j < EDGEDIM) accE += a * edge_attr[(size_t)e * EDGEDIM + j];
    }
    __bf16* f = feats + ((size_t)n * H4 + h) * FPAD;
    float2 xn = ((const float2*)x)[(size_t)n * 64 + j];
    f[2 * j]           = (__bf16)xn.x;
    f[2 * j + 1]       = (__bf16)xn.y;
    f[INDIM + 2 * j]     = (__bf16)acc0;
    f[INDIM + 2 * j + 1] = (__bf16)acc1;
    if (j < EDGEDIM) {
        f[2 * INDIM + j] = (__bf16)accE;
        f[FMLP + j]      = (__bf16)0.f;   // zero pad 272..287
    }
}

// ---------- 7: MLP layer 1 via WMMA bf16, B staged in LDS, 2 M-tiles/wave ----------
// grid: (ceil(nTiles/16), H4); block 256 = 8 waves, all same head, 2 tiles each
__global__ __launch_bounds__(256) void mlp1_kernel(
    const __bf16* __restrict__ feats, const __bf16* __restrict__ w1bf,
    const float* __restrict__ pb1, float* __restrict__ h1,
    float* __restrict__ bn_sum, float* __restrict__ bn_sq, int N) {
    __shared__ __bf16 sB[OUTH * LDS1_STRIDE];   // ~37 KB
    __shared__ float s_sum[OUTH];
    __shared__ float s_sq[OUTH];
    int tid = threadIdx.x;
    int head = blockIdx.y;
    if (tid < OUTH) { s_sum[tid] = 0.f; s_sq[tid] = 0.f; }
    {   // stage this head's weights: 64 rows x 288 bf16, padded row stride
        const __bf16* wsrc = w1bf + (size_t)head * OUTH * FPAD;
        for (int i = tid; i < OUTH * (FPAD / 8); i += 256) {
            int r = i / (FPAD / 8);
            int ch = i % (FPAD / 8);
            *(v8bf*)&sB[r * LDS1_STRIDE + ch * 8] =
                ((const v8bf*)(wsrc + (size_t)r * FPAD))[ch];
        }
    }
    __syncthreads();

    int lane = tid & 31;
    int wv = tid >> 5;
    int m = lane & 15;
    int hi = lane >> 4;
    int nTiles = (N + 15) >> 4;
    int tile0 = blockIdx.x * 16 + wv * 2;

    const __bf16* arow[2];
    int rowbase[2];
#pragma unroll
    for (int mt = 0; mt < 2; ++mt) {
        int tile = tile0 + mt;
        rowbase[mt] = tile << 4;
        int rowA = rowbase[mt] + m;
        if (rowA >= N) rowA = N - 1;              // clamp (wave-uniform tail)
        arow[mt] = feats + ((size_t)rowA * H4 + head) * FPAD;
    }

    v8f c[2][4];
#pragma unroll
    for (int mt = 0; mt < 2; ++mt)
#pragma unroll
        for (int ot = 0; ot < 4; ++ot) c[mt][ot] = v8f{};

#pragma unroll
    for (int kb = 0; kb < FPAD; kb += 32) {
        v16bf a[2];
#pragma unroll
        for (int mt = 0; mt < 2; ++mt) {
            const __bf16* ap = arow[mt] + kb + hi * 8;
            v8bf alo = *(const v8bf*)ap;
            v8bf ahi = *(const v8bf*)(ap + 16);
#pragma unroll
            for (int i = 0; i < 8; ++i) { a[mt][i] = alo[i]; a[mt][8 + i] = ahi[i]; }
        }
#pragma unroll
        for (int ot = 0; ot < 4; ++ot) {
            const __bf16* bp = &sB[(ot * 16 + m) * LDS1_STRIDE + kb + hi * 8];
            v8bf blo = *(const v8bf*)bp;
            v8bf bhi = *(const v8bf*)(bp + 16);
            v16bf b;
#pragma unroll
            for (int i = 0; i < 8; ++i) { b[i] = blo[i]; b[8 + i] = bhi[i]; }
            c[0][ot] = __builtin_amdgcn_wmma_f32_16x16x32_bf16(
                false, a[0], false, b, (short)0, c[0][ot], false, false);
            c[1][ot] = __builtin_amdgcn_wmma_f32_16x16x32_bf16(
                false, a[1], false, b, (short)0, c[1][ot], false, false);
        }
    }

#pragma unroll
    for (int ot = 0; ot < 4; ++ot) {
        int o = ot * 16 + m;
        float bias = pb1[head * OUTH + o];
        float sum = 0.f, sq = 0.f;
#pragma unroll
        for (int mt = 0; mt < 2; ++mt) {
#pragma unroll
            for (int r = 0; r < 8; ++r) {
                int rowO = rowbase[mt] + r + 8 * hi;
                float v = c[mt][ot][r] + bias;
                v = v > 0.f ? v : 0.f;
                if (rowO < N) {
                    h1[((size_t)rowO * H4 + head) * OUTH + o] = v;
                    sum += v; sq += v * v;
                }
            }
        }
        sum += __shfl_xor(sum, 16, 32);
        sq  += __shfl_xor(sq, 16, 32);
        if (hi == 0) {
            atomicAdd(&s_sum[o], sum);
            atomicAdd(&s_sq[o], sq);
        }
    }
    __syncthreads();
    if (tid < OUTH) {
        atomicAdd(&bn_sum[head * OUTH + tid], s_sum[tid]);
        atomicAdd(&bn_sq[head * OUTH + tid],  s_sq[tid]);
    }
}

// ---------- 8: finalize BN: fold to per-channel scale/shift ----------
__global__ void bn_finalize_kernel(const float* __restrict__ bn_sum,
                                   const float* __restrict__ bn_sq,
                                   const float* __restrict__ gamma,
                                   const float* __restrict__ beta,
                                   float* __restrict__ scaleA,
                                   float* __restrict__ shiftB, int N) {
    int i = threadIdx.x;
    if (i < 256) {
        float invN = 1.f / (float)N;
        float mu = bn_sum[i] * invN;
        float var = bn_sq[i] * invN - mu * mu;
        float sc = gamma[i] * rsqrtf(var + 1e-5f);
        scaleA[i] = sc;
        shiftB[i] = beta[i] - mu * sc;
    }
}

// ---------- 9: MLP layer 2 via WMMA bf16 (BN folded into A), B in LDS ----------
__global__ __launch_bounds__(256) void mlp2_kernel(
    const float* __restrict__ h1, const float* __restrict__ scaleA,
    const float* __restrict__ shiftB, const __bf16* __restrict__ w2bf,
    const float* __restrict__ pb2, float* __restrict__ out, int N) {
    __shared__ __bf16 sB[OUTH * LDS2_STRIDE];   // 9 KB
    __shared__ float s_sc[OUTH];
    __shared__ float s_sh[OUTH];
    int tid = threadIdx.x;
    int head = blockIdx.y;
    if (tid < OUTH) {
        s_sc[tid] = scaleA[head * OUTH + tid];
        s_sh[tid] = shiftB[head * OUTH + tid];
    }
    {
        const __bf16* wsrc = w2bf + (size_t)head * OUTH * OUTH;
        for (int i = tid; i < OUTH * (OUTH / 8); i += 256) {
            int r = i / (OUTH / 8);
            int ch = i % (OUTH / 8);
            *(v8bf*)&sB[r * LDS2_STRIDE + ch * 8] =
                ((const v8bf*)(wsrc + (size_t)r * OUTH))[ch];
        }
    }
    __syncthreads();

    int lane = tid & 31;
    int wv = tid >> 5;
    int m = lane & 15;
    int hi = lane >> 4;
    int nTiles = (N + 15) >> 4;
    int tile0 = blockIdx.x * 16 + wv * 2;

    const float* arow[2];
    int rowbase[2];
#pragma unroll
    for (int mt = 0; mt < 2; ++mt) {
        int tile = tile0 + mt;
        rowbase[mt] = tile << 4;
        int rowA = rowbase[mt] + m;
        if (rowA >= N) rowA = N - 1;
        arow[mt] = h1 + ((size_t)rowA * H4 + head) * OUTH;
    }

    v8f c[2][4];
#pragma unroll
    for (int mt = 0; mt < 2; ++mt)
#pragma unroll
        for (int ot = 0; ot < 4; ++ot) c[mt][ot] = v8f{};

#pragma unroll
    for (int kb = 0; kb < OUTH; kb += 32) {
        v16bf a[2];
#pragma unroll
        for (int mt = 0; mt < 2; ++mt) {
            int k0 = kb + hi * 8;
            float4 f0 = *(const float4*)(arow[mt] + k0);
            float4 f1 = *(const float4*)(arow[mt] + k0 + 4);
            float4 f2 = *(const float4*)(arow[mt] + k0 + 16);
            float4 f3 = *(const float4*)(arow[mt] + k0 + 20);
            float va[16] = {f0.x, f0.y, f0.z, f0.w, f1.x, f1.y, f1.z, f1.w,
                            f2.x, f2.y, f2.z, f2.w, f3.x, f3.y, f3.z, f3.w};
#pragma unroll
            for (int i = 0; i < 8; ++i) {
                int ka = k0 + i, kbb = k0 + 16 + i;
                a[mt][i]     = (__bf16)(va[i]     * s_sc[ka]  + s_sh[ka]);
                a[mt][8 + i] = (__bf16)(va[8 + i] * s_sc[kbb] + s_sh[kbb]);
            }
        }
#pragma unroll
        for (int ot = 0; ot < 4; ++ot) {
            const __bf16* bp = &sB[(ot * 16 + m) * LDS2_STRIDE + kb + hi * 8];
            v8bf blo = *(const v8bf*)bp;
            v8bf bhi = *(const v8bf*)(bp + 16);
            v16bf b;
#pragma unroll
            for (int i = 0; i < 8; ++i) { b[i] = blo[i]; b[8 + i] = bhi[i]; }
            c[0][ot] = __builtin_amdgcn_wmma_f32_16x16x32_bf16(
                false, a[0], false, b, (short)0, c[0][ot], false, false);
            c[1][ot] = __builtin_amdgcn_wmma_f32_16x16x32_bf16(
                false, a[1], false, b, (short)0, c[1][ot], false, false);
        }
    }

#pragma unroll
    for (int ot = 0; ot < 4; ++ot) {
        int p = ot * 16 + m;
        float bias = pb2[head * OUTH + p];
#pragma unroll
        for (int mt = 0; mt < 2; ++mt) {
#pragma unroll
            for (int r = 0; r < 8; ++r) {
                int rowO = rowbase[mt] + r + 8 * hi;
                if (rowO < N)
                    out[(size_t)rowO * (H4 * OUTH) + head * OUTH + p] = c[mt][ot][r] + bias;
            }
        }
    }
}

// ---------- launch ----------
extern "C" void kernel_launch(void* const* d_in, const int* in_sizes, int n_in,
                              void* d_out, int out_size, void* d_ws, size_t ws_size,
                              hipStream_t stream) {
    const float* x         = (const float*)d_in[0];
    const int*   row       = (const int*)d_in[1];
    const int*   col       = (const int*)d_in[2];
    const float* edge_attr = (const float*)d_in[3];
    const float* a1w       = (const float*)d_in[4];
    const float* a2w       = (const float*)d_in[5];
    const float* aew       = (const float*)d_in[6];
    const float* w1        = (const float*)d_in[7];
    const float* b1        = (const float*)d_in[8];
    const float* gamma     = (const float*)d_in[9];
    const float* beta      = (const float*)d_in[10];
    const float* w2        = (const float*)d_in[11];
    const float* b2        = (const float*)d_in[12];
    float* out = (float*)d_out;

    const int N = in_sizes[0] / INDIM;
    const int E = in_sizes[1];

    // workspace carve-out (256B aligned)
    char* p = (char*)d_ws;
    auto carve = [&](size_t bytes) -> void* {
        void* r = (void*)p;
        p += (bytes + 255) & ~(size_t)255;
        return r;
    };
    float*    exbuf = (float*)carve((size_t)E * H4 * 4);
    unsigned* mkey  = (unsigned*)carve((size_t)N * H4 * 4);
    float*    s     = (float*)carve((size_t)N * H4 * 4);
    float*    attn1 = (float*)carve((size_t)N * H4 * 4);
    float*    attn2 = (float*)carve((size_t)N * H4 * 4);
    int*      rp    = (int*)carve((size_t)(N + 1) * 4);
    __bf16*   feats = (__bf16*)carve((size_t)N * H4 * FPAD * 2);
    float*    h1    = (float*)carve((size_t)N * H4 * OUTH * 4);
    __bf16*   w1bf  = (__bf16*)carve((size_t)H4 * OUTH * FPAD * 2);
    __bf16*   w2bf  = (__bf16*)carve((size_t)H4 * OUTH * OUTH * 2);
    float*    bnbuf = (float*)carve(1024 * 4);
    float* bn_sum = bnbuf;
    float* bn_sq  = bnbuf + 256;
    float* scaleA = bnbuf + 512;
    float* shiftB = bnbuf + 768;

    const int nh = N * H4;
    init_kernel<<<(nh + 255) / 256, 256, 0, stream>>>(mkey, s, bnbuf, nh);
    node_logits_kernel<<<(N + 7) / 8, 256, 0, stream>>>(x, a1w, a2w, attn1, attn2, N);
    {
        int tot = H4 * OUTH * FPAD + H4 * OUTH * OUTH;
        pack_weights_kernel<<<(tot + 255) / 256, 256, 0, stream>>>(w1, w2, w1bf, w2bf);
    }
    edge_logits_kernel<<<(E + 255) / 256, 256, 0, stream>>>(
        row, col, edge_attr, aew, attn1, attn2, exbuf, mkey, E);
    edge_exp_kernel<<<(E + 255) / 256, 256, 0, stream>>>(row, exbuf, mkey, s, E);
    row_ptr_kernel<<<(N + 1 + 255) / 256, 256, 0, stream>>>(row, rp, N, E);
    aggregate_kernel<<<N, 256, 0, stream>>>(x, col, edge_attr, exbuf, s, rp, feats, N);
    {
        int nTiles = (N + 15) / 16;
        dim3 grid((nTiles + 15) / 16, H4);
        mlp1_kernel<<<grid, 256, 0, stream>>>(feats, w1bf, b1, h1, bn_sum, bn_sq, N);
        bn_finalize_kernel<<<1, 256, 0, stream>>>(bn_sum, bn_sq, gamma, beta, scaleA, shiftB, N);
        mlp2_kernel<<<grid, 256, 0, stream>>>(h1, scaleA, shiftB, w2bf, b2, out, N);
    }
}
